// RGCNN_model_4982162063585
// MI455X (gfx1250) — matrix operand order repaired and checked by reading, not compiled, not run on hardware
//
#include <hip/hip_runtime.h>
#include <hip/hip_bf16.h>
#include <math.h>

// MI455X / gfx1250, wave32.
// - Gram/dist, Tx@W GEMMs, h^T(Lh), MLP head: V_WMMA_F32_16X16X4_F32 (exact f32)
// - Dominant Adj@X spmm: V_WMMA_F32_16X16X32_BF16 (bf16 in, f32 acc, 8x matrix
//   rate, half Adj HBM traffic). B operand produced in fragment-native packed
//   layout by the D^-1/2 column-scale kernel -> both fragments are b128 loads.
// - Tile count NT is a template parameter so every WMMA chain is statically
//   unrolled; A fragment is double-buffered one k-step ahead; sched_group
//   barriers group the VMEM reads ahead of the matrix ops; global_prefetch
//   streams the Adj rows.

typedef __attribute__((ext_vector_type(2))) float v2f;
typedef __attribute__((ext_vector_type(8))) float v8f;
typedef __attribute__((ext_vector_type(4))) unsigned int v4u;
typedef __attribute__((ext_vector_type(8))) unsigned int v8u;
typedef __attribute__((ext_vector_type(16))) __bf16 v16bf;

#define BATCH 16
#define NPTS  2048
#define BNTOT (BATCH * NPTS)

static __device__ __forceinline__ v8f wmma_f32_k4(v2f a, v2f b, v8f c) {
  return __builtin_amdgcn_wmma_f32_16x16x4_f32(false, a, false, b, (short)0, c,
                                               false, false);
}

static __device__ __forceinline__ v8f vzero8() {
  v8f z = {0.f, 0.f, 0.f, 0.f, 0.f, 0.f, 0.f, 0.f};
  return z;
}

static __device__ __forceinline__ unsigned short f2bf(float f) {
  unsigned int u = __float_as_uint(f);
  u += 0x7FFFu + ((u >> 16) & 1u);  // round-to-nearest-even
  return (unsigned short)(u >> 16);
}

// ---------------------------------------------------------------- utilities

__global__ void fill_kernel(float* __restrict__ p, float v, int n) {
  int i = blockIdx.x * blockDim.x + threadIdx.x;
  if (i < n) p[i] = v;
}

__global__ void pad_x_kernel(const float* __restrict__ x, float* __restrict__ xp) {
  int i = blockIdx.x * blockDim.x + threadIdx.x;
  if (i >= BNTOT * 16) return;
  int n = i >> 4, c = i & 15;
  xp[i] = (c < 6) ? x[n * 6 + c] : 0.f;
}

__global__ void pad_w1_kernel(const float* __restrict__ w, float* __restrict__ wp) {
  int i = blockIdx.x * blockDim.x + threadIdx.x;
  const int tot = 6 * 16 * 128;
  if (i >= tot) return;
  int k = i / (16 * 128);
  int rem = i % (16 * 128);
  int c = rem / 128, f = rem % 128;
  wp[i] = (c < 6) ? w[(k * 6 + c) * 128 + f] : 0.f;
}

__global__ void sqnorm_kernel(const float* __restrict__ X, float* __restrict__ sq,
                              int Dk) {
  int i = blockIdx.x * blockDim.x + threadIdx.x;
  if (i >= BNTOT) return;
  const float* xr = X + (size_t)i * Dk;
  float s = 0.f;
  for (int c = 0; c < Dk; ++c) { float v = xr[c]; s += v * v; }
  sq[i] = s;
}

__global__ void dinv_kernel(const float* __restrict__ deg, float* __restrict__ dinv,
                            int n) {
  int i = blockIdx.x * blockDim.x + threadIdx.x;
  if (i >= n) return;
  float d = deg[i];
  dinv[i] = (d > 0.f) ? rsqrtf(d) : 0.f;
}

// Fused D^-1/2 column scale + transpose + bf16 pair-pack into the WMMA
// B-fragment-native layout: SPt[b][col][p] = {bf16(s[2p,col]), bf16(s[2p+1,col])}
__global__ void colscale_pack_kernel(const float* __restrict__ T,
                                     const float* __restrict__ dinv,
                                     unsigned int* __restrict__ SPt, int F,
                                     long long n) {
  long long idx = (long long)blockIdx.x * blockDim.x + threadIdx.x;
  if (idx >= n) return;  // n = BATCH * F * (NPTS/2)
  const int NP2 = NPTS / 2;
  long long per_b = (long long)F * NP2;
  int b = (int)(idx / per_b);
  long long rem = idx - (long long)b * per_b;
  int col = (int)(rem / NP2);
  int pr = (int)(rem % NP2);
  int n0 = 2 * pr, n1 = 2 * pr + 1;
  const float* Tb = T + (size_t)b * NPTS * F;
  const float* dv = dinv + (size_t)b * NPTS;
  float s0 = dv[n0] * Tb[(size_t)n0 * F + col];
  float s1 = dv[n1] * Tb[(size_t)n1 * F + col];
  SPt[idx] = (unsigned int)f2bf(s0) | ((unsigned int)f2bf(s1) << 16);
}

// --------------------------------------------- adjacency: W = exp(-d2), deg
__global__ void adj_kernel(const float* __restrict__ X, const float* __restrict__ sq,
                           unsigned short* __restrict__ Adj, float* __restrict__ deg,
                           int Dk) {
  const int N = NPTS;
  int lane = threadIdx.x & 31;
  int wid = threadIdx.x >> 5;
  int hf = lane >> 4, r = lane & 15;
  int b = blockIdx.z;
  int i = blockIdx.x * 16;
  int jb = (blockIdx.y * 4 + wid) * 64;
  const float* Xb = X + (size_t)b * N * Dk;
  const float* sqb = sq + (size_t)b * N;
  unsigned short* Ab = Adj + (size_t)b * N * N;
  float* degb = deg + (size_t)b * N;

  v8f acc[4] = {vzero8(), vzero8(), vzero8(), vzero8()};
  int kofs = 2 * hf;
  int arow = i + r;
  for (int k = 0; k < Dk; k += 4) {
    v2f a = *(const v2f*)(Xb + (size_t)arow * Dk + k + kofs);
    v2f b0 = *(const v2f*)(Xb + (size_t)(jb + r) * Dk + k + kofs);
    v2f b1 = *(const v2f*)(Xb + (size_t)(jb + 16 + r) * Dk + k + kofs);
    v2f b2 = *(const v2f*)(Xb + (size_t)(jb + 32 + r) * Dk + k + kofs);
    v2f b3 = *(const v2f*)(Xb + (size_t)(jb + 48 + r) * Dk + k + kofs);
    acc[0] = wmma_f32_k4(a, b0, acc[0]);
    acc[1] = wmma_f32_k4(a, b1, acc[1]);
    acc[2] = wmma_f32_k4(a, b2, acc[2]);
    acc[3] = wmma_f32_k4(a, b3, acc[3]);
    __builtin_amdgcn_sched_group_barrier(0x020, 5, 0);  // 5 loads first
    __builtin_amdgcn_sched_group_barrier(0x008, 4, 0);  // then 4 matrix ops
  }

  float rowsum[8];
#pragma unroll
  for (int v = 0; v < 8; ++v) rowsum[v] = 0.f;
#pragma unroll
  for (int t = 0; t < 4; ++t) {
    int n = jb + t * 16 + r;
    float sj = sqb[n];
#pragma unroll
    for (int v = 0; v < 8; ++v) {
      int m = i + v + 8 * hf;
      float d2 = sqb[m] + sj - 2.f * acc[t][v];
      float w = (m == n) ? 0.f : __expf(-d2);
      Ab[(size_t)m * N + n] = f2bf(w);
      rowsum[v] += w;
    }
  }
#pragma unroll
  for (int v = 0; v < 8; ++v) {
    float s = rowsum[v];
    s += __shfl_xor(s, 1, 16);
    s += __shfl_xor(s, 2, 16);
    s += __shfl_xor(s, 4, 16);
    s += __shfl_xor(s, 8, 16);
    if (r == 0) atomicAdd(&degb[i + v + 8 * hf], s);
  }
}

// ---------------- Y = alpha * (-dinv_i) * (Adj @ S) + beta * Z  in bf16 WMMA
template <int NT>
__global__ void spmm_bf16_kernel(const unsigned short* __restrict__ Adj,
                                 const unsigned int* __restrict__ SPt,
                                 const float* __restrict__ dinv,
                                 const float* __restrict__ Z,
                                 float* __restrict__ Y, int F, float alpha,
                                 float beta) {
  const int N = NPTS;
  int lane = threadIdx.x & 31;
  int wid = threadIdx.x >> 5;
  int hf = lane >> 4, r = lane & 15;
  int b = blockIdx.z;
  int i = blockIdx.x * 16;
  int jb = (blockIdx.y * 4 + wid) * (16 * NT);
  if (jb + 16 * NT > F) return;  // wave-uniform

  const unsigned short* Ab =
      Adj + (size_t)b * N * N + (size_t)(i + r) * N + 8 * hf;
  const unsigned int* Sb = SPt + (size_t)b * F * (N / 2);
  const float* Zb = Z + (size_t)b * N * F;
  const float* dvb = dinv + (size_t)b * N;
  float* Yb = Y + (size_t)b * N * F;

  const unsigned int* Bp[NT];
#pragma unroll
  for (int t = 0; t < NT; ++t)
    Bp[t] = Sb + (size_t)(jb + 16 * t + r) * (N / 2) + 4 * hf;

  v8f acc[NT];
#pragma unroll
  for (int t = 0; t < NT; ++t) acc[t] = vzero8();

  // software pipeline: A fragment buffered one k-step ahead
  v4u a0 = *(const v4u*)(Ab);
  v4u a1 = *(const v4u*)(Ab + 16);
  for (int k0 = 0; k0 < N; k0 += 32) {
    v8u araw = {a0.x, a0.y, a0.z, a0.w, a1.x, a1.y, a1.z, a1.w};
    v16bf afrag = __builtin_bit_cast(v16bf, araw);
    v4u b0[NT], b1[NT];
#pragma unroll
    for (int t = 0; t < NT; ++t) {
      b0[t] = *(const v4u*)(Bp[t] + (k0 >> 1));
      b1[t] = *(const v4u*)(Bp[t] + (k0 >> 1) + 8);
    }
    if (k0 + 32 < N) {  // uniform
      a0 = *(const v4u*)(Ab + k0 + 32);
      a1 = *(const v4u*)(Ab + k0 + 48);
      __builtin_prefetch(Ab + k0 + 160, 0, 0);  // stream Adj ~256B ahead
    }
#pragma unroll
    for (int t = 0; t < NT; ++t) {
      v8u braw = {b0[t].x, b0[t].y, b0[t].z, b0[t].w,
                  b1[t].x, b1[t].y, b1[t].z, b1[t].w};
      v16bf bfrag = __builtin_bit_cast(v16bf, braw);
      acc[t] = __builtin_amdgcn_wmma_f32_16x16x32_bf16(
          false, afrag, false, bfrag, (short)0, acc[t], false, false);
    }
    // group this iteration's VMEM reads ahead of its matrix ops
    __builtin_amdgcn_sched_group_barrier(0x020, 2 * NT + 2, 0);
    __builtin_amdgcn_sched_group_barrier(0x008, NT, 0);
  }

  bool useZ = (beta != 0.f);
#pragma unroll
  for (int t = 0; t < NT; ++t) {
    int col = jb + t * 16 + r;
#pragma unroll
    for (int v = 0; v < 8; ++v) {
      int m = i + v + 8 * hf;
      float val = -alpha * dvb[m] * acc[t][v];
      if (useZ) val += beta * Zb[(size_t)m * F + col];
      Yb[(size_t)m * F + col] = val;
    }
  }
}

// ------------------------------------ C (+)= A[BN,K] @ W[K,F]  (+bias,+relu)
template <int NT>
__global__ void gemm_tw_kernel(const float* __restrict__ A, const float* __restrict__ W,
                               const float* __restrict__ bias, float* __restrict__ C,
                               int K, int F, int accumulate, int addBias,
                               int applyRelu) {
  int lane = threadIdx.x & 31;
  int wid = threadIdx.x >> 5;
  int hf = lane >> 4, r = lane & 15;
  int i = blockIdx.x * 16;
  int jb = (blockIdx.y * 4 + wid) * (16 * NT);
  if (jb + 16 * NT > F) return;

  v8f acc[NT];
#pragma unroll
  for (int t = 0; t < NT; ++t) acc[t] = vzero8();
  int kofs = 2 * hf;
  int arow = i + r;
  for (int k = 0; k < K; k += 4) {
    v2f a = *(const v2f*)(A + (size_t)arow * K + k + kofs);
    v2f bf[NT];
#pragma unroll
    for (int t = 0; t < NT; ++t) {
      int col = jb + t * 16 + r;
      bf[t].x = W[(size_t)(k + kofs) * F + col];
      bf[t].y = W[(size_t)(k + kofs + 1) * F + col];
    }
#pragma unroll
    for (int t = 0; t < NT; ++t) acc[t] = wmma_f32_k4(a, bf[t], acc[t]);
    __builtin_amdgcn_sched_group_barrier(0x020, 2 * NT + 1, 0);
    __builtin_amdgcn_sched_group_barrier(0x008, NT, 0);
  }
#pragma unroll
  for (int t = 0; t < NT; ++t) {
    int col = jb + t * 16 + r;
#pragma unroll
    for (int v = 0; v < 8; ++v) {
      int m = i + v + 8 * hf;
      size_t off = (size_t)m * F + col;
      float val = acc[t][v];
      if (addBias) val += bias[col];
      if (accumulate) val += C[off];
      if (applyRelu) val = fmaxf(val, 0.f);
      C[off] = val;
    }
  }
}

// --------------------------------------------- M[F,F] = H^T @ LH over K=BN
template <int NT>
__global__ void gram_ft_kernel(const float* __restrict__ H,
                               const float* __restrict__ LH,
                               float* __restrict__ M, int Ktot, int F) {
  int lane = threadIdx.x & 31;
  int wid = threadIdx.x >> 5;
  int hf = lane >> 4, r = lane & 15;
  int i = blockIdx.x * 16;
  int jb = (blockIdx.y * 4 + wid) * (16 * NT);
  if (jb + 16 * NT > F) return;

  v8f acc[NT];
#pragma unroll
  for (int t = 0; t < NT; ++t) acc[t] = vzero8();
  int kofs = 2 * hf;
  for (int k = 0; k < Ktot; k += 4) {
    v2f a;
    a.x = H[(size_t)(k + kofs) * F + i + r];
    a.y = H[(size_t)(k + kofs + 1) * F + i + r];
    v2f bf[NT];
#pragma unroll
    for (int t = 0; t < NT; ++t) {
      int col = jb + t * 16 + r;
      bf[t].x = LH[(size_t)(k + kofs) * F + col];
      bf[t].y = LH[(size_t)(k + kofs + 1) * F + col];
    }
#pragma unroll
    for (int t = 0; t < NT; ++t) acc[t] = wmma_f32_k4(a, bf[t], acc[t]);
    __builtin_amdgcn_sched_group_barrier(0x020, 2 * NT + 2, 0);
    __builtin_amdgcn_sched_group_barrier(0x008, NT, 0);
  }
#pragma unroll
  for (int t = 0; t < NT; ++t) {
    int col = jb + t * 16 + r;
#pragma unroll
    for (int v = 0; v < 8; ++v) {
      int m = i + v + 8 * hf;
      M[(size_t)m * F + col] = acc[t][v];
    }
  }
}

// ---------------------------- C[16,Nout] = A[16,K] @ Wt[Nout,K]^T (MLP head)
__global__ void gemm_nt_kernel(const float* __restrict__ A, const float* __restrict__ Wt,
                               const float* __restrict__ bias, float* __restrict__ C,
                               int K, int Nout, int applyRelu) {
  int lane = threadIdx.x & 31;
  int wid = threadIdx.x >> 5;
  int hf = lane >> 4, r = lane & 15;
  int jb = (blockIdx.y * 4 + wid) * 64;
  if (jb >= Nout) return;
  int nt = (Nout - jb) >= 64 ? 4 : (Nout - jb + 15) / 16;

  v8f acc[4] = {vzero8(), vzero8(), vzero8(), vzero8()};
  int kofs = 2 * hf;
  for (int k = 0; k < K; k += 4) {
    v2f a = *(const v2f*)(A + (size_t)r * K + k + kofs);
    for (int t = 0; t < nt; ++t) {
      int wrow = jb + t * 16 + r;
      v2f bf;
      if (wrow < Nout) {
        bf = *(const v2f*)(Wt + (size_t)wrow * K + k + kofs);
      } else {
        bf.x = 0.f; bf.y = 0.f;
      }
      acc[t] = wmma_f32_k4(a, bf, acc[t]);
    }
  }
  for (int t = 0; t < nt; ++t) {
    int col = jb + t * 16 + r;
    if (col < Nout) {
#pragma unroll
      for (int v = 0; v < 8; ++v) {
        int m = v + 8 * hf;
        float val = acc[t][v] + bias[col];
        if (applyRelu) val = fmaxf(val, 0.f);
        C[(size_t)m * Nout + col] = val;
      }
    }
  }
}

__global__ void maxpool_kernel(const float* __restrict__ H, float* __restrict__ G,
                               int F) {
  int i = blockIdx.x * blockDim.x + threadIdx.x;
  if (i >= BATCH * F) return;
  int b = i / F, f = i % F;
  const float* Hb = H + (size_t)b * NPTS * F + f;
  float m = -3.4e38f;
  for (int n = 0; n < NPTS; ++n) m = fmaxf(m, Hb[(size_t)n * F]);
  G[i] = m;
}

__global__ void sumsq_kernel(const float* __restrict__ X, long long n,
                             float* __restrict__ acc) {
  long long idx = (long long)blockIdx.x * blockDim.x + threadIdx.x;
  long long stride = (long long)gridDim.x * blockDim.x;
  float s = 0.f;
  for (long long i = idx; i < n; i += stride) s += X[i] * X[i];
  for (int o = 16; o >= 1; o >>= 1) s += __shfl_xor(s, o, 32);
  if ((threadIdx.x & 31) == 0) atomicAdd(acc, s);
}

__global__ void finalize_regs_kernel(const float* __restrict__ acc,
                                     float* __restrict__ out) {
  out[0] = sqrtf(acc[0]);
  out[1] = sqrtf(acc[1]);
  out[2] = sqrtf(acc[2]);
  float nw = sqrtf(acc[3]);
  float nb = sqrtf(acc[4]);
  out[3] = nw; out[4] = nb;
  out[5] = nw; out[6] = nb;
  out[7] = nw; out[8] = nb;
}

// ---------------------------------------------------------------- launcher

extern "C" void kernel_launch(void* const* d_in, const int* in_sizes, int n_in,
                              void* d_out, int out_size, void* d_ws, size_t ws_size,
                              hipStream_t stream) {
  const float* x   = (const float*)d_in[0];
  const float* w1  = (const float*)d_in[4];
  const float* b1  = (const float*)d_in[5];
  const float* w2  = (const float*)d_in[6];
  const float* b2  = (const float*)d_in[7];
  const float* w3  = (const float*)d_in[8];
  const float* b3  = (const float*)d_in[9];
  const float* fw1 = (const float*)d_in[10];
  const float* fb1 = (const float*)d_in[11];
  const float* fw2 = (const float*)d_in[12];
  const float* fb2 = (const float*)d_in[13];
  const float* fw3 = (const float*)d_in[14];
  const float* fb3 = (const float*)d_in[15];
  float* out = (float*)d_out;

  // workspace carve (~0.8 GB)
  float* p = (float*)d_ws;
  auto take = [&](size_t nf) { float* r = p; p += nf; return r; };
  unsigned short* AdjB =
      (unsigned short*)take((size_t)BATCH * NPTS * NPTS / 2);  // 128 MB bf16
  unsigned int* SPt =
      (unsigned int*)take((size_t)BATCH * 1024 * (NPTS / 2));  // 64 MB packed
  float* HA   = take((size_t)BNTOT * 1024);                    // 128 MB
  float* HB   = take((size_t)BNTOT * 1024);                    // 128 MB
  float* T1   = take((size_t)BNTOT * 512);                     // 64 MB
  float* T2   = take((size_t)BNTOT * 512);
  float* T3   = take((size_t)BNTOT * 512);
  float* LH   = take((size_t)BNTOT * 1024);                    // 128 MB
  float* Xp   = take((size_t)BNTOT * 16);
  float* W1p  = take((size_t)6 * 16 * 128);
  float* sqb  = take((size_t)BNTOT);
  float* deg  = take((size_t)BNTOT);
  float* dinv = take((size_t)BNTOT);
  float* Mmat = take((size_t)1024 * 1024);
  float* G    = take((size_t)BATCH * 1024);
  float* O1   = take((size_t)BATCH * 512);
  float* O2   = take((size_t)BATCH * 128);
  float* accS = take((size_t)8);

  auto cdiv = [](long long a, long long b) { return (int)((a + b - 1) / b); };

  fill_kernel<<<1, 32, 0, stream>>>(accS, 0.f, 8);
  pad_x_kernel<<<cdiv(BNTOT * 16, 256), 256, 0, stream>>>(x, Xp);
  pad_w1_kernel<<<cdiv(6 * 16 * 128, 256), 256, 0, stream>>>(w1, W1p);

  auto spmm = [&](const float* Tsrc, const float* Z, float* Y, int F, float al,
                  float be) {
    long long nPack = (long long)BATCH * F * (NPTS / 2);
    colscale_pack_kernel<<<cdiv(nPack, 256), 256, 0, stream>>>(Tsrc, dinv, SPt, F,
                                                               nPack);
    if (F >= 64) {
      dim3 g(NPTS / 16, cdiv(F, 256), BATCH);
      spmm_bf16_kernel<4><<<g, 128, 0, stream>>>(AdjB, SPt, dinv, Z, Y, F, al, be);
    } else {
      dim3 g(NPTS / 16, cdiv(F, 64), BATCH);
      spmm_bf16_kernel<1><<<g, 128, 0, stream>>>(AdjB, SPt, dinv, Z, Y, F, al, be);
    }
  };

  auto layer = [&](const float* hin, int Fin, const float* W, const float* bias,
                   int Kc, int Fout, float* hout, int regIdx) {
    // graph from hin
    sqnorm_kernel<<<cdiv(BNTOT, 256), 256, 0, stream>>>(hin, sqb, Fin);
    fill_kernel<<<cdiv(BNTOT, 256), 256, 0, stream>>>(deg, 0.f, BNTOT);
    adj_kernel<<<dim3(NPTS / 16, NPTS / 256, BATCH), 128, 0, stream>>>(hin, sqb,
                                                                       AdjB, deg,
                                                                       Fin);
    dinv_kernel<<<cdiv(BNTOT, 256), 256, 0, stream>>>(deg, dinv, BNTOT);

    dim3 gW(BNTOT / 16, cdiv(Fout, 256));

    // out = Tx0 @ W[0] + bias
    gemm_tw_kernel<4><<<gW, 128, 0, stream>>>(hin, W, bias, hout, Fin, Fout, 0, 1,
                                              0);

    // Tx1 = Lhat @ Tx0 ; out += Tx1 @ W[1]
    const float* tx0 = hin;
    float* tx1 = T1;
    spmm(tx0, tx0, tx1, Fin, 1.f, 0.f);
    gemm_tw_kernel<4><<<gW, 128, 0, stream>>>(tx1, W + (size_t)Fin * Fout, bias,
                                              hout, Fin, Fout, 1, 0,
                                              (Kc == 2) ? 1 : 0);

    // Tx2 = 2*Lhat@Tx1 - Tx0 ; out += Tx2 @ W[k]  (relu fused into last term)
    float* rot[3] = {T2, T3, T1};
    for (int k = 2; k < Kc; ++k) {
      float* tx2 = rot[(k - 2) % 3];
      spmm(tx1, tx0, tx2, Fin, 2.f, -1.f);
      gemm_tw_kernel<4><<<gW, 128, 0, stream>>>(tx2, W + (size_t)k * Fin * Fout,
                                                bias, hout, Fin, Fout, 1, 0,
                                                (k == Kc - 1) ? 1 : 0);
      tx0 = tx1;
      tx1 = tx2;
    }

    // regularizer: LH = (I + Lhat) h ; M = h^T LH ; reg = ||M||_F
    spmm(hout, hout, LH, Fout, 1.f, 1.f);
    gram_ft_kernel<4><<<dim3(Fout / 16, cdiv(Fout, 256)), 128, 0, stream>>>(
        hout, LH, Mmat, BNTOT, Fout);
    sumsq_kernel<<<256, 256, 0, stream>>>(Mmat, (long long)Fout * Fout,
                                          accS + regIdx);
  };

  layer(Xp, 16, W1p, b1, 6, 128, HA, 0);
  layer(HA, 128, w2, b2, 5, 512, HB, 1);
  layer(HB, 512, w3, b3, 3, 1024, HA, 2);

  // global max pool + MLP head
  maxpool_kernel<<<cdiv(BATCH * 1024, 256), 256, 0, stream>>>(HA, G, 1024);
  gemm_nt_kernel<<<dim3(1, cdiv(512, 256)), 128, 0, stream>>>(G, fw1, fb1, O1, 1024,
                                                              512, 1);
  gemm_nt_kernel<<<dim3(1, 1), 128, 0, stream>>>(O1, fw2, fb2, O2, 512, 128, 1);
  gemm_nt_kernel<<<dim3(1, 1), 128, 0, stream>>>(O2, fw3, fb3, out, 128, 10, 0);

  // parameter norms + finalize regs into d_out[160..168]
  sumsq_kernel<<<256, 256, 0, stream>>>(fw1, (long long)512 * 1024, accS + 3);
  sumsq_kernel<<<16, 256, 0, stream>>>(fb1, (long long)512, accS + 4);
  finalize_regs_kernel<<<1, 1, 0, stream>>>(accS, out + 160);
}
